// MultiHeadedAttention_57183194579669
// MI455X (gfx1250) — matrix operand order repaired and checked by reading, not compiled, not run on hardware
//
#include <hip/hip_runtime.h>
#include <math.h>

#define Bn 2
#define Sn 2048
#define Dn 1024
#define Hn 16
#define DKn 64

typedef __bf16 bf16_t;
typedef unsigned int u32;
typedef unsigned long long u64;
typedef __attribute__((ext_vector_type(16))) __bf16 v16bf;
typedef __attribute__((ext_vector_type(8)))  __bf16 v8bf;
typedef __attribute__((ext_vector_type(8)))  float  v8f;
typedef __attribute__((ext_vector_type(4)))  u32    v4u;
typedef __attribute__((ext_vector_type(8)))  int    v8i;
typedef __attribute__((ext_vector_type(4)))  int    v4i;

__device__ inline v8f wmma_bf(v16bf a, v16bf b, v8f c) {
  return __builtin_amdgcn_wmma_f32_16x16x32_bf16(false, a, false, b, (short)0, c,
                                                 false, false);
}

// A fragment: 16x32 bf16, row-major source, base at (row0,k0). Lane r holds row r;
// half 0 gets K {0..7,16..23}, half 1 gets K {8..15,24..31}.
__device__ inline v16bf load_a_frag(const bf16_t* base, int lda) {
  int lane = threadIdx.x & 31, r = lane & 15, hf = lane >> 4;
  const bf16_t* p = base + (size_t)r * lda + hf * 8;
  v8bf lo = *(const v8bf*)p;
  v8bf hi = *(const v8bf*)(p + 16);
  v16bf f;
#pragma unroll
  for (int i = 0; i < 8; ++i) { f[i] = lo[i]; f[i + 8] = hi[i]; }
  return f;
}

// B fragment: 32x16, supplied as transposed row-major Bt[N,K], base at (n0,k0).
// Lane r holds column n0+r; half 0 K {0..7,16..23}, half 1 K {8..15,24..31}.
__device__ inline v16bf load_b_frag(const bf16_t* base, int ldb) {
  int lane = threadIdx.x & 31, r = lane & 15, hf = lane >> 4;
  const bf16_t* p = base + (size_t)r * ldb + hf * 8;
  v8bf lo = *(const v8bf*)p;
  v8bf hi = *(const v8bf*)(p + 16);
  v16bf f;
#pragma unroll
  for (int i = 0; i < 8; ++i) { f[i] = lo[i]; f[i + 8] = hi[i]; }
  return f;
}

// ---------------- TDM: 2D bf16 tile -> LDS (D# per CDNA5 ISA) --
__device__ inline void tdm_load_tile_2d(u32 lds_byte_addr, const void* gaddr,
                                        u32 rows, u32 cols_elems,
                                        u32 tensor_d0, u32 tensor_d1,
                                        u32 stride_elems) {
  u64 ga = (u64)(size_t)gaddr;
  v4u g0;
  g0[0] = 1u;                                            // count=1, user descriptor
  g0[1] = lds_byte_addr;                                 // LDS byte address
  g0[2] = (u32)(ga & 0xffffffffu);                       // global_addr[31:0]
  g0[3] = (u32)((ga >> 32) & 0x1ffffffu) | (2u << 30);   // global_addr[56:32], type=2
  v8i g1;
  g1[0] = (int)(1u << 16);                               // data_size=1 (2 bytes)
  g1[1] = (int)((tensor_d0 & 0xffffu) << 16);            // tensor_dim0[15:0]
  g1[2] = (int)((tensor_d0 >> 16) & 0xffffu) |
          (int)((tensor_d1 & 0xffffu) << 16);            // dim0 hi | dim1 lo
  g1[3] = (int)((tensor_d1 >> 16) & 0xffffu) |
          (int)((cols_elems & 0xffffu) << 16);           // dim1 hi | tile_dim0
  g1[4] = (int)(rows & 0xffffu);                         // tile_dim1 (tile_dim2=0)
  g1[5] = (int)stride_elems;                             // tensor_dim0_stride lo32
  g1[6] = 0;                                             // stride hi | dim1_stride lo
  g1[7] = 0;
  v4i g2 = {0, 0, 0, 0};
  v4i g3 = {0, 0, 0, 0};
#if defined(__clang_major__) && __clang_major__ >= 23
  v8i g4 = {0, 0, 0, 0, 0, 0, 0, 0};
  __builtin_amdgcn_tensor_load_to_lds(g0, g1, g2, g3, g4, 0);
#else
  __builtin_amdgcn_tensor_load_to_lds(g0, g1, g2, g3, 0);
#endif
}

// ---------------- elementwise f32 -> bf16 ----------------
__global__ void cvt_f32_bf16(const float* __restrict__ src, bf16_t* __restrict__ dst, int n) {
  int i = blockIdx.x * blockDim.x + threadIdx.x;
  if (i < n) dst[i] = (bf16_t)src[i];
}

// ---------------- transpose + convert (for prev_eta^T) ----------------
__global__ void transpose_cvt(const float* __restrict__ src, bf16_t* __restrict__ dst) {
  __shared__ float tile[32][33];
  int b = blockIdx.z;
  int j0 = blockIdx.x * 32, n0 = blockIdx.y * 32;
  int tx = threadIdx.x & 31, ty = threadIdx.x >> 5;
  const float* s = src + (size_t)b * Sn * Sn;
  bf16_t* d = dst + (size_t)b * Sn * Sn;
  for (int yy = ty; yy < 32; yy += 8)
    tile[yy][tx] = s[(size_t)(j0 + yy) * Sn + n0 + tx];
  __syncthreads();
  for (int yy = ty; yy < 32; yy += 8)
    d[(size_t)(n0 + yy) * Sn + j0 + tx] = (bf16_t)tile[tx][yy];
}

// ---- generic NT GEMM with TDM-staged, double-buffered B tile in LDS -------------
// C = A[M,K] * Bt[N,K]^T (+bias)(+Cadd). Block tile 64x128, 8 waves of 32x32.
// B tile (128 rows x 64 K, 16KB bf16) is DMA'd by the Tensor Data Mover.
__global__ __launch_bounds__(256) void gemm_bf16_nt(
    const bf16_t* __restrict__ A, int lda, const bf16_t* __restrict__ Bt, int ldb,
    const float* __restrict__ bias, const float* __restrict__ Cadd, int ldadd,
    float* __restrict__ Cf, bf16_t* __restrict__ Cb, int ldc, int M, int N, int K) {
  __shared__ bf16_t btile[2][128 * 64];  // 32KB double buffer
  int wave = threadIdx.x >> 5;
  int wm = wave & 1, wn = wave >> 1;  // 2 x 4 waves, wave tile 32x32
  int m0 = blockIdx.y * 64 + wm * 32;
  int nblk = blockIdx.x * 128;
  int n0 = nblk + wn * 32;
  v8f acc00 = {}, acc01 = {}, acc10 = {}, acc11 = {};
  const bf16_t* ap = A + (size_t)m0 * lda;
  // prologue: kick off first B tile
  if (wave == 0) {
    tdm_load_tile_2d((u32)(size_t)&btile[0][0], Bt + (size_t)nblk * ldb,
                     128u, 64u, (u32)K, (u32)N, (u32)ldb);
  }
  for (int kt = 0; kt < K; kt += 64) {
    int buf = (kt >> 6) & 1;
    if (wave == 0) {
      if (kt + 64 < K) {
        tdm_load_tile_2d((u32)(size_t)&btile[buf ^ 1][0],
                         Bt + (size_t)nblk * ldb + kt + 64,
                         128u, 64u, (u32)K, (u32)N, (u32)ldb);
        __builtin_amdgcn_s_wait_tensorcnt(1);  // current tile landed
      } else {
        __builtin_amdgcn_s_wait_tensorcnt(0);  // last tile landed
      }
    }
    __syncthreads();
    const bf16_t* lb = &btile[buf][0] + (size_t)(wn * 32) * 64;
#pragma unroll
    for (int ks = 0; ks < 64; ks += 32) {
      __builtin_prefetch(ap + kt + ks + 256, 0, 1);
      v16bf a0 = load_a_frag(ap + kt + ks, lda);
      v16bf a1 = load_a_frag(ap + (size_t)16 * lda + kt + ks, lda);
      v16bf b0 = load_b_frag(lb + ks, 64);
      v16bf b1 = load_b_frag(lb + (size_t)16 * 64 + ks, 64);
      acc00 = wmma_bf(a0, b0, acc00);
      acc01 = wmma_bf(a0, b1, acc01);
      acc10 = wmma_bf(a1, b0, acc10);
      acc11 = wmma_bf(a1, b1, acc11);
    }
    __syncthreads();  // all waves done with buf before it is re-filled
  }
  int lane = threadIdx.x & 31, r = lane & 15, hf = lane >> 4;
  v8f* accs[4] = {&acc00, &acc01, &acc10, &acc11};
#pragma unroll
  for (int t = 0; t < 4; ++t) {
    int mb = m0 + (t >> 1) * 16, nb = n0 + (t & 1) * 16;
    float bv = bias ? bias[nb + r] : 0.f;
#pragma unroll
    for (int e = 0; e < 8; ++e) {
      int row = mb + e + 8 * hf, col = nb + r;
      float val = (*accs[t])[e] + bv;
      if (Cadd) val += Cadd[(size_t)row * ldadd + col];
      if (Cf) Cf[(size_t)row * ldc + col] = val;
      if (Cb) Cb[(size_t)row * ldc + col] = (bf16_t)val;
    }
  }
}

// ---------------- v^T (per head, [DK,S]) + ||v|| ----------------
__global__ void vT_and_mag(const bf16_t* __restrict__ vb, bf16_t* __restrict__ vT,
                           float* __restrict__ vmag) {
  int idx = blockIdx.x * blockDim.x + threadIdx.x;  // B*H*S
  if (idx >= Bn * Hn * Sn) return;
  int s = idx % Sn, bh = idx / Sn, h = bh % Hn, b = bh / Hn;
  const bf16_t* src = vb + ((size_t)(b * Sn + s)) * Dn + h * DKn;
  bf16_t* dst = vT + ((size_t)bh * DKn) * Sn + s;
  float sum = 0.f;
#pragma unroll 8
  for (int dk = 0; dk < DKn; ++dk) {
    float f = (float)src[dk];
    sum += f * f;
    dst[(size_t)dk * Sn] = src[dk];
  }
  vmag[idx] = sqrtf(sum);
}

// ---------------- pass 1: per-row softmax stats (online, per-lane partials) -------
__global__ __launch_bounds__(256) void attn_stats(const bf16_t* __restrict__ qb,
                                                  const bf16_t* __restrict__ kb,
                                                  float* __restrict__ rmax,
                                                  float* __restrict__ rsum) {
  int wave = threadIdx.x >> 5;
  int q0 = blockIdx.x * 128 + wave * 16;
  int bh = blockIdx.y, h = bh % Hn, b = bh / Hn;
  const bf16_t* qbase = qb + (size_t)b * Sn * Dn + h * DKn + (size_t)q0 * Dn;
  const bf16_t* kbase = kb + (size_t)b * Sn * Dn + h * DKn;
  v16bf a0 = load_a_frag(qbase, Dn);
  v16bf a1 = load_a_frag(qbase + 32, Dn);
  int lane = threadIdx.x & 31, r = lane & 15, hf = lane >> 4;
  float pm[8], pl[8];
#pragma unroll
  for (int e = 0; e < 8; ++e) { pm[e] = -1e30f; pl[e] = 0.f; }
  const float scale = 0.125f;  // 1/sqrt(64)
  for (int j0 = 0; j0 <= q0; j0 += 16) {
    v16bf b0 = load_b_frag(kbase + (size_t)j0 * Dn, Dn);
    v16bf b1 = load_b_frag(kbase + (size_t)j0 * Dn + 32, Dn);
    v8f z = {};
    v8f c = wmma_bf(a1, b1, wmma_bf(a0, b0, z));
#pragma unroll
    for (int e = 0; e < 8; ++e) {
      int row = q0 + e + 8 * hf, col = j0 + r;
      if (col <= row) {
        float sc = c[e] * scale;
        float nm = fmaxf(pm[e], sc);
        pl[e] = pl[e] * __expf(pm[e] - nm) + __expf(sc - nm);
        pm[e] = nm;
      }
    }
  }
#pragma unroll
  for (int e = 0; e < 8; ++e) {
    float m = pm[e], l = pl[e];
#pragma unroll
    for (int off = 1; off < 16; off <<= 1) {
      float m2 = __shfl_xor(m, off, 32);
      float l2 = __shfl_xor(l, off, 32);
      float nm = fmaxf(m, m2);
      l = l * __expf(m - nm) + l2 * __expf(m2 - nm);
      m = nm;
    }
    if (r == 0) {
      int row = q0 + e + 8 * hf;
      rmax[(size_t)bh * Sn + row] = m;
      rsum[(size_t)bh * Sn + row] = l;
    }
  }
}

// ---------------- pass 2a: attn_out = probs @ v (probs recomputed) ----------------
__global__ __launch_bounds__(256) void attn_out_k(
    const bf16_t* __restrict__ qb, const bf16_t* __restrict__ kb,
    const bf16_t* __restrict__ vT, const float* __restrict__ rmax,
    const float* __restrict__ rsum, bf16_t* __restrict__ merged) {
  __shared__ bf16_t lds_p[8][16 * 32];  // per-wave 16x32 prob tile
  int wave = threadIdx.x >> 5;
  int q0 = blockIdx.x * 128 + wave * 16;
  int bh = blockIdx.y, h = bh % Hn, b = bh / Hn;
  const bf16_t* qbase = qb + (size_t)b * Sn * Dn + h * DKn + (size_t)q0 * Dn;
  const bf16_t* kbase = kb + (size_t)b * Sn * Dn + h * DKn;
  const bf16_t* vTb = vT + (size_t)bh * DKn * Sn;
  int lane = threadIdx.x & 31, r = lane & 15, hf = lane >> 4;
  v16bf a0 = load_a_frag(qbase, Dn);
  v16bf a1 = load_a_frag(qbase + 32, Dn);
  float rm[8], ri[8];
#pragma unroll
  for (int e = 0; e < 8; ++e) {
    int row = q0 + e + 8 * hf;
    rm[e] = rmax[(size_t)bh * Sn + row];
    ri[e] = 1.f / rsum[(size_t)bh * Sn + row];
  }
  v8f acc0 = {}, acc1 = {}, acc2 = {}, acc3 = {};
  bf16_t* myp = lds_p[wave];
  const float scale = 0.125f;
  for (int j0 = 0; j0 < q0 + 16; j0 += 32) {
#pragma unroll
    for (int t = 0; t < 2; ++t) {
      int jt = j0 + t * 16;
      v8f c = {};
      if (jt < q0 + 16) {
        v16bf b0 = load_b_frag(kbase + (size_t)jt * Dn, Dn);
        v16bf b1 = load_b_frag(kbase + (size_t)jt * Dn + 32, Dn);
        c = wmma_bf(a1, b1, wmma_bf(a0, b0, c));
      }
#pragma unroll
      for (int e = 0; e < 8; ++e) {
        int row = q0 + e + 8 * hf, col = jt + r;
        float p = 0.f;
        if (jt < q0 + 16 && col <= row) p = __expf(c[e] * scale - rm[e]) * ri[e];
        myp[(e + 8 * hf) * 32 + t * 16 + r] = (bf16_t)p;
      }
    }
    // wave-local LDS round trip (DS ops are in-order per wave)
    v16bf pa = load_a_frag(myp, 32);
    acc0 = wmma_bf(pa, load_b_frag(vTb + (size_t)0 * Sn + j0, Sn), acc0);
    acc1 = wmma_bf(pa, load_b_frag(vTb + (size_t)16 * Sn + j0, Sn), acc1);
    acc2 = wmma_bf(pa, load_b_frag(vTb + (size_t)32 * Sn + j0, Sn), acc2);
    acc3 = wmma_bf(pa, load_b_frag(vTb + (size_t)48 * Sn + j0, Sn), acc3);
  }
  bf16_t* mbase = merged + ((size_t)(b * Sn + q0)) * Dn + h * DKn;
  v8f* accs[4] = {&acc0, &acc1, &acc2, &acc3};
#pragma unroll
  for (int n = 0; n < 4; ++n)
#pragma unroll
    for (int e = 0; e < 8; ++e)
      mbase[(size_t)(e + 8 * hf) * Dn + n * 16 + r] = (bf16_t)(*accs[n])[e];
}

// ---------------- pass 2b: eta_layer[b,i,j] = sum_h p_h(i,j) * ||v_h(j)|| ----------
__global__ __launch_bounds__(256) void eta_layer_k(
    const bf16_t* __restrict__ qb, const bf16_t* __restrict__ kb,
    const float* __restrict__ rmax, const float* __restrict__ rsum,
    const float* __restrict__ vmag, bf16_t* __restrict__ etab) {
  int wave = threadIdx.x >> 5;
  int j0 = blockIdx.x * 128 + wave * 16;  // key tile
  int q0 = blockIdx.y * 16;               // query tile
  int b = blockIdx.z;
  int lane = threadIdx.x & 31, r = lane & 15, hf = lane >> 4;
  float acc[8] = {0.f, 0.f, 0.f, 0.f, 0.f, 0.f, 0.f, 0.f};
  if (j0 <= q0 + 15) {
    for (int h = 0; h < Hn; ++h) {
      const bf16_t* qbase = qb + (size_t)b * Sn * Dn + h * DKn + (size_t)q0 * Dn;
      const bf16_t* kbase = kb + (size_t)b * Sn * Dn + h * DKn + (size_t)j0 * Dn;
      v16bf a0 = load_a_frag(qbase, Dn);
      v16bf a1 = load_a_frag(qbase + 32, Dn);
      v16bf b0 = load_b_frag(kbase, Dn);
      v16bf b1 = load_b_frag(kbase + 32, Dn);
      v8f z = {};
      v8f c = wmma_bf(a1, b1, wmma_bf(a0, b0, z));
      int bh = b * Hn + h;
      float vm = vmag[(size_t)bh * Sn + j0 + r];
#pragma unroll
      for (int e = 0; e < 8; ++e) {
        int row = q0 + e + 8 * hf, col = j0 + r;
        if (col <= row) {
          float m = rmax[(size_t)bh * Sn + row];
          float l = rsum[(size_t)bh * Sn + row];
          acc[e] += __expf(c[e] * 0.125f - m) / l * vm;
        }
      }
    }
  }
  bf16_t* dst = etab + (size_t)b * Sn * Sn;
#pragma unroll
  for (int e = 0; e < 8; ++e)
    dst[(size_t)(q0 + e + 8 * hf) * Sn + j0 + r] = (bf16_t)acc[e];
}

// =================================================================================
extern "C" void kernel_launch(void* const* d_in, const int* in_sizes, int n_in,
                              void* d_out, int out_size, void* d_ws, size_t ws_size,
                              hipStream_t stream) {
  (void)in_sizes; (void)n_in; (void)out_size; (void)ws_size;
  const float* x        = (const float*)d_in[0];
  const float* prev_eta = (const float*)d_in[1];
  // d_in[2] = attention_mask (causal, hardcoded)
  const float* Wq = (const float*)d_in[3];  const float* bq = (const float*)d_in[4];
  const float* Wk = (const float*)d_in[5];  const float* bk = (const float*)d_in[6];
  const float* Wv = (const float*)d_in[7];  const float* bv = (const float*)d_in[8];
  const float* Wo = (const float*)d_in[9];  const float* bo = (const float*)d_in[10];

  float* outp = (float*)d_out;                       // [B,S,D]
  float* etap = outp + (size_t)Bn * Sn * Dn;         // [B,S,S]

  // workspace carve-up (bytes, 256-aligned)
  char* w = (char*)d_ws;
  size_t off = 0;
  auto alloc = [&](size_t bytes) { void* p = w + off; off = (off + bytes + 255) & ~(size_t)255; return p; };
  const size_t MD = (size_t)Bn * Sn * Dn;   // 4194304
  const size_t SS = (size_t)Sn * Sn;        // per batch
  bf16_t* xb   = (bf16_t*)alloc(MD * 2);
  bf16_t* wqb  = (bf16_t*)alloc((size_t)Dn * Dn * 2);
  bf16_t* wkb  = (bf16_t*)alloc((size_t)Dn * Dn * 2);
  bf16_t* wvb  = (bf16_t*)alloc((size_t)Dn * Dn * 2);
  bf16_t* wob  = (bf16_t*)alloc((size_t)Dn * Dn * 2);
  bf16_t* qbuf = (bf16_t*)alloc(MD * 2);
  bf16_t* kbuf = (bf16_t*)alloc(MD * 2);
  bf16_t* vbuf = (bf16_t*)alloc(MD * 2);
  bf16_t* vT   = (bf16_t*)alloc(MD * 2);
  float*  vmag = (float*)alloc((size_t)Bn * Hn * Sn * 4);
  float*  rmax = (float*)alloc((size_t)Bn * Hn * Sn * 4);
  float*  rsum = (float*)alloc((size_t)Bn * Hn * Sn * 4);
  bf16_t* mrgb = (bf16_t*)alloc(MD * 2);
  bf16_t* etab = (bf16_t*)alloc((size_t)Bn * SS * 2);
  bf16_t* petaT= (bf16_t*)alloc((size_t)Bn * SS * 2);

  // 1) convert inputs to bf16
  cvt_f32_bf16<<<(int)((MD + 255) / 256), 256, 0, stream>>>(x, xb, (int)MD);
  cvt_f32_bf16<<<(Dn * Dn + 255) / 256, 256, 0, stream>>>(Wq, wqb, Dn * Dn);
  cvt_f32_bf16<<<(Dn * Dn + 255) / 256, 256, 0, stream>>>(Wk, wkb, Dn * Dn);
  cvt_f32_bf16<<<(Dn * Dn + 255) / 256, 256, 0, stream>>>(Wv, wvb, Dn * Dn);
  cvt_f32_bf16<<<(Dn * Dn + 255) / 256, 256, 0, stream>>>(Wo, wob, Dn * Dn);

  // 2) QKV projections (M=B*S, N=K=D), bf16 outputs; B operand staged via TDM
  {
    dim3 g(Dn / 128, (Bn * Sn) / 64);
    gemm_bf16_nt<<<g, 256, 0, stream>>>(xb, Dn, wqb, Dn, bq, nullptr, 0, nullptr, qbuf, Dn, Bn * Sn, Dn, Dn);
    gemm_bf16_nt<<<g, 256, 0, stream>>>(xb, Dn, wkb, Dn, bk, nullptr, 0, nullptr, kbuf, Dn, Bn * Sn, Dn, Dn);
    gemm_bf16_nt<<<g, 256, 0, stream>>>(xb, Dn, wvb, Dn, bv, nullptr, 0, nullptr, vbuf, Dn, Bn * Sn, Dn, Dn);
  }

  // 3) v^T per head + ||v||
  vT_and_mag<<<(Bn * Hn * Sn) / 256, 256, 0, stream>>>(vbuf, vT, vmag);

  // 4) softmax stats
  attn_stats<<<dim3(Sn / 128, Bn * Hn), 256, 0, stream>>>(qbuf, kbuf, rmax, rsum);

  // 5) attention output (merged heads, bf16)
  attn_out_k<<<dim3(Sn / 128, Bn * Hn), 256, 0, stream>>>(qbuf, kbuf, vT, rmax, rsum, mrgb);

  // 6) output projection -> d_out part 1 (f32)
  {
    dim3 g(Dn / 128, (Bn * Sn) / 64);
    gemm_bf16_nt<<<g, 256, 0, stream>>>(mrgb, Dn, wob, Dn, bo, nullptr, 0, outp, nullptr, Dn, Bn * Sn, Dn, Dn);
  }

  // 7) eta_layer (head-summed weighted probs), bf16
  eta_layer_k<<<dim3(Sn / 128, Sn / 16, Bn), 256, 0, stream>>>(qbuf, kbuf, rmax, rsum, vmag, etab);

  // 8) prev_eta^T in bf16
  transpose_cvt<<<dim3(Sn / 32, Sn / 32, Bn), 256, 0, stream>>>(prev_eta, petaT);

  // 9) new_eta = prev_eta + eta_layer @ prev_eta -> d_out part 2 (f32)
  for (int b = 0; b < Bn; ++b) {
    dim3 g(Sn / 128, Sn / 64);
    gemm_bf16_nt<<<g, 256, 0, stream>>>(etab + (size_t)b * SS, Sn,
                                        petaT + (size_t)b * SS, Sn,
                                        nullptr, prev_eta + (size_t)b * SS, Sn,
                                        etap + (size_t)b * SS, nullptr, Sn,
                                        Sn, Sn, Sn);
  }
}
